// Att_RNN_GRU_45251775430854
// MI455X (gfx1250) — matrix operand order repaired and checked by reading, not compiled, not run on hardware
//
#include <hip/hip_runtime.h>
#include <math.h>
#include <stdint.h>

// Problem constants (from reference)
constexpr int CB = 128;    // batch
constexpr int CT = 1024;   // time
constexpr int CI = 128;    // input dim
constexpr int CH = 256;    // hidden
constexpr int CG = 768;    // 3*H
constexpr int CA = 40;     // attention dim
constexpr int NWG_SCAN = 16;   // persistent workgroups in the scan (H/16 j-tiles)

typedef __attribute__((ext_vector_type(16))) __bf16 v16bf;
typedef __attribute__((ext_vector_type(8)))  float  v8f;

// ---------------------------------------------------------------------------
// Fragment builders for CDNA5 wave32 WMMA; A and B share one contiguous k order (not the ISA table's)
// A (16x32 bf16): lanes 0-15 -> rows 0-15, K k0+0..15
//                 lanes 16-31 -> rows 0-15, K k0+16..31 (contiguous per lane)
// B (32x16 bf16): lane -> col (lane&15); lanes 0-15 hold K k0+0..15,
//                 lanes 16-31 hold K k0+16..31 (contiguous per lane)
// C/D (16x16 f32): VGPR v, lane l -> row v + 8*(l>>4), col l&15
// ---------------------------------------------------------------------------

__device__ inline v16bf frag_a_from_f32(const float* base, int ldm, int row0,
                                        int k0, int lane) {
  int r  = row0 + (lane & 15);
  int kA = k0 + ((lane >> 4) ? 16 : 0);
  const float* p0 = base + (size_t)r * ldm + kA;        // K kA..kA+7
  const float* p1 = p0 + 8;                             // K kA+8..kA+15
  v16bf a;
#pragma unroll
  for (int i = 0; i < 8; ++i) {
    a[i]     = (__bf16)p0[i];
    a[8 + i] = (__bf16)p1[i];
  }
  return a;
}

__device__ inline v16bf frag_a_from_bf16(const __bf16* base, int ldm, int row0,
                                         int k0, int lane) {
  int r  = row0 + (lane & 15);
  int kA = k0 + ((lane >> 4) ? 16 : 0);
  const __bf16* p0 = base + (size_t)r * ldm + kA;
  v16bf a;
#pragma unroll
  for (int i = 0; i < 8; ++i) {
    a[i]     = p0[i];
    a[8 + i] = p0[8 + i];
  }
  return a;
}

// B from row-major [N,K] f32 weights (row n = output column n)
__device__ inline v16bf frag_b_from_f32(const float* base, int ldn, int n0,
                                        int k0, int lane) {
  const float* p = base + (size_t)(n0 + (lane & 15)) * ldn + k0 + ((lane >> 4) << 4);
  v16bf b;
#pragma unroll
  for (int i = 0; i < 16; ++i) b[i] = (__bf16)p[i];
  return b;
}

// B from bf16 LDS tile [16 rows][ldk]
__device__ inline v16bf frag_b_from_lds(const __bf16* lds, int ldk, int k0,
                                        int lane) {
  const __bf16* p = lds + (size_t)(lane & 15) * ldk + k0 + ((lane >> 4) << 4);
  v16bf b;
#pragma unroll
  for (int i = 0; i < 16; ++i) b[i] = p[i];
  return b;
}

__device__ inline v8f wmma_bf16(v16bf a, v16bf b, v8f c) {
  return __builtin_amdgcn_wmma_f32_16x16x32_bf16(
      /*neg_a=*/false, a, /*neg_b=*/false, b,
      /*c_mod=*/(short)0, c, /*reuse_a=*/false, /*reuse_b=*/false);
}

// ---------------------------------------------------------------------------
// Tensor Data Mover: async-load the per-step gx tile into LDS.
// Tile is 3D: 16 contiguous cols (x) * 3 gates (y, stride 256 elems)
//            * 128 batch rows (z, stride T*3H elems), data_size = 2 bytes.
// LDS fill order is x-then-y-then-z => linear [(row*3 + gate)*16 + col].
// ---------------------------------------------------------------------------
#if __has_builtin(__builtin_amdgcn_tensor_load_to_lds)
#define HAVE_TDM 1
typedef unsigned int u32x4 __attribute__((ext_vector_type(4)));
typedef int          i32x4 __attribute__((ext_vector_type(4)));
typedef int          i32x8 __attribute__((ext_vector_type(8)));

__device__ inline void tdm_load_gx_tile(const __bf16* gsrc, unsigned lds_byte_addr) {
  unsigned long long ga = (unsigned long long)(uintptr_t)gsrc;
  const unsigned d1s = (unsigned)CT * (unsigned)CG;   // tensor_dim1_stride = 786432
  u32x4 g0;
  g0[0] = 1u;                                          // count=1 (valid, user mode)
  g0[1] = lds_byte_addr;                               // lds_addr (bytes)
  g0[2] = (unsigned)(ga & 0xFFFFFFFFu);                // global_addr[31:0]
  g0[3] = (unsigned)((ga >> 32) & 0x01FFFFFFu) | (2u << 30);  // addr[56:32] | type=2
  i32x8 g1;
  g1[0] = (1 << 16);                                   // data_size = 2 bytes
  g1[1] = (16 << 16);                                  // tensor_dim0 = 16 (lo16 in hi half)
  g1[2] = (3 << 16);                                   // dim0 hi=0 | tensor_dim1 = 3
  g1[3] = (16 << 16);                                  // dim1 hi=0 | tile_dim0 = 16
  g1[4] = 3 | (128 << 16);                             // tile_dim1 = 3 | tile_dim2 = 128
  g1[5] = 256;                                         // tensor_dim0_stride[31:0]
  g1[6] = (int)((d1s & 0xFFFFu) << 16);                // d0s[47:32]=0 | d1s[15:0]
  g1[7] = (int)(d1s >> 16);                            // tensor_dim1_stride[47:16]
  i32x4 g2;
  g2[0] = 128;                                         // tensor_dim2 = 128
  g2[1] = 0; g2[2] = 0; g2[3] = 0;                     // no dim3, iterate off
  i32x4 g3; g3[0] = 0; g3[1] = 0; g3[2] = 0; g3[3] = 0;
#if defined(__clang_major__) && __clang_major__ >= 23
  i32x8 gz; 
#pragma unroll
  for (int i = 0; i < 8; ++i) gz[i] = 0;
  __builtin_amdgcn_tensor_load_to_lds(g0, g1, g2, g3, gz, 0);
#else
  __builtin_amdgcn_tensor_load_to_lds(g0, g1, g2, g3, 0);
#endif
}
#else
#define HAVE_TDM 0
#endif

// ---------------------------------------------------------------------------
// Kernel 1: gx[b*T+t, g] = x[b*T+t, :] . W_ih[g, :] + b_ih[g]   (bf16 out)
// M = B*T = 131072, N = 768, K = 128. One wave per 16x16 tile, 8 waves/block.
// ---------------------------------------------------------------------------
__global__ void __launch_bounds__(256)
k_gemm_gx(const float* __restrict__ x, const float* __restrict__ W_ih,
          const float* __restrict__ b_ih, __bf16* __restrict__ gx) {
  const int lane = threadIdx.x & 31;
  const int wave = threadIdx.x >> 5;
  const int tile = blockIdx.x * 8 + wave;
  const int ntiles_n = CG / 16;             // 48
  const int nt = tile % ntiles_n;
  const int mt = tile / ntiles_n;
  const int row0 = mt * 16;
  const int n0 = nt * 16;

  const float bias = b_ih[n0 + (lane & 15)];
  v8f c;
#pragma unroll
  for (int v = 0; v < 8; ++v) c[v] = bias;

#pragma unroll
  for (int k0 = 0; k0 < CI; k0 += 32) {
    v16bf a = frag_a_from_f32(x, CI, row0, k0, lane);
    v16bf b = frag_b_from_f32(W_ih, CI, n0, k0, lane);
    c = wmma_bf16(a, b, c);
  }

#pragma unroll
  for (int v = 0; v < 8; ++v) {
    int r  = row0 + v + ((lane >> 4) << 3);
    int cc = n0 + (lane & 15);
    gx[(size_t)r * CG + cc] = (__bf16)c[v];
  }
}

// ---------------------------------------------------------------------------
// Kernel 2: init h0 = 0 and barrier counter = 0 (must run every call).
// ---------------------------------------------------------------------------
__global__ void k_scan_init(__bf16* __restrict__ hbuf, unsigned* __restrict__ bar) {
  int idx = blockIdx.x * blockDim.x + threadIdx.x;
  if (idx == 0) *bar = 0u;
  for (int i = idx; i < CB * CH; i += gridDim.x * blockDim.x)
    hbuf[i] = (__bf16)0.0f;
}

// ---------------------------------------------------------------------------
// Kernel 3: persistent GRU scan. 16 WGs (j-tile each), 8 waves (b-tile each).
// Per step: 3 gate GEMM tiles (K=256 -> 8 WMMAs each) + fused gate math,
// gx staged via TDM (double-buffered LDS, issued one step ahead),
// then a device-wide barrier.
// ---------------------------------------------------------------------------
__device__ inline void grid_barrier(unsigned* bar, unsigned target) {
  __threadfence();
  __syncthreads();
  if (threadIdx.x == 0) {
    __hip_atomic_fetch_add(bar, 1u, __ATOMIC_ACQ_REL, __HIP_MEMORY_SCOPE_AGENT);
    while (__hip_atomic_load(bar, __ATOMIC_ACQUIRE, __HIP_MEMORY_SCOPE_AGENT) < target)
      __builtin_amdgcn_s_sleep(1);
  }
  __syncthreads();
}

__global__ void __launch_bounds__(256)
k_gru_scan(const __bf16* __restrict__ gx, __bf16* __restrict__ outs,
           __bf16* __restrict__ hbuf, unsigned* __restrict__ bar,
           const float* __restrict__ W_hh, const float* __restrict__ b_hh) {
  __shared__ __bf16 sW[3][16][CH];         // 24KB: this WG's 16 cols of each gate
#if HAVE_TDM
  __shared__ __bf16 sGx[2][CB * 48];       // 24KB: double-buffered per-step gx tile
#endif

  const int tid  = threadIdx.x;
  const int lane = tid & 31;
  const int wave = tid >> 5;
  const int jt   = blockIdx.x;       // j-tile: columns jt*16 .. jt*16+15
  const int jc0  = jt * 16;
  const int bRow0 = wave * 16;       // batch rows bRow0 .. bRow0+15

#if HAVE_TDM
  // Kick off the DMA for step 0 immediately (gx is ready: kernel boundary).
  if (wave == 0) {
    tdm_load_gx_tile(gx + (size_t)0 * CG + jc0,
                     (unsigned)(uintptr_t)&sGx[0][0]);
  }
#endif

  // Preload W_hh slices (rows {g*256 + jc0 + rr}, all K) as bf16 into LDS.
  for (int i = tid; i < 3 * 16 * CH; i += 256) {
    int g = i / (16 * CH);
    int rr = (i / CH) & 15;
    int c = i % CH;
    sW[g][rr][c] = (__bf16)W_hh[(size_t)(g * CH + jc0 + rr) * CH + c];
  }
  __syncthreads();

  const int j = jc0 + (lane & 15);
  const float biasR = b_hh[j];
  const float biasZ = b_hh[CH + j];
  const float biasN = b_hh[2 * CH + j];

  for (int t = 0; t < CT; ++t) {
    const __bf16* hc = hbuf + ((t & 1) ? (size_t)CB * CH : 0);
    __bf16* hn = hbuf + ((t & 1) ? 0 : (size_t)CB * CH);

#if !HAVE_TDM
    // prefetch next step's gx slab for this lane's first row
    if (t + 1 < CT) {
      size_t pf = ((size_t)bRow0 * CT + (t + 1)) * CG + j;
      __builtin_prefetch(gx + pf, 0, 1);
    }
#endif

    v8f aR, aZ, aN;
#pragma unroll
    for (int v = 0; v < 8; ++v) { aR[v] = biasR; aZ[v] = biasZ; aN[v] = biasN; }

#pragma unroll
    for (int k0 = 0; k0 < CH; k0 += 32) {
      v16bf a  = frag_a_from_bf16(hc, CH, bRow0, k0, lane);
      v16bf bR = frag_b_from_lds(&sW[0][0][0], CH, k0, lane);
      v16bf bZ = frag_b_from_lds(&sW[1][0][0], CH, k0, lane);
      v16bf bN = frag_b_from_lds(&sW[2][0][0], CH, k0, lane);
      aR = wmma_bf16(a, bR, aR);
      aZ = wmma_bf16(a, bZ, aZ);
      aN = wmma_bf16(a, bN, aN);
    }

#if HAVE_TDM
    // Make this step's DMA'd gx tile visible to all waves.
    if (wave == 0) __builtin_amdgcn_s_wait_tensorcnt(0);
    __syncthreads();
    const __bf16* gxl = &sGx[t & 1][0];
#endif

    // Fused GRU gate math on the 16x16 tile (this lane owns col j, 8 rows).
#pragma unroll
    for (int v = 0; v < 8; ++v) {
      int row = bRow0 + v + ((lane >> 4) << 3);
#if HAVE_TDM
      int li = (row * 3) * 16 + (lane & 15);
      float xr = (float)gxl[li];
      float xz = (float)gxl[li + 16];
      float xn = (float)gxl[li + 32];
#else
      size_t gxo = ((size_t)row * CT + t) * CG + j;
      float xr = (float)gx[gxo];
      float xz = (float)gx[gxo + CH];
      float xn = (float)gx[gxo + 2 * CH];
#endif
      float hold = (float)hc[(size_t)row * CH + j];
      float r = 1.0f / (1.0f + __expf(-(xr + aR[v])));
      float z = 1.0f / (1.0f + __expf(-(xz + aZ[v])));
      float n = tanhf(xn + r * aN[v]);
      float hv = (1.0f - z) * n + z * hold;
      hn[(size_t)row * CH + j] = (__bf16)hv;
      outs[((size_t)row * CT + t) * CH + j] = (__bf16)hv;
    }

#if HAVE_TDM
    // Overlap next step's gx DMA with the barrier + next WMMA chain.
    if (t + 1 < CT && wave == 0) {
      tdm_load_gx_tile(gx + (size_t)(t + 1) * CG + jc0,
                       (unsigned)(uintptr_t)&sGx[(t + 1) & 1][0]);
    }
#endif

    grid_barrier(bar, (unsigned)NWG_SCAN * (unsigned)(t + 1));
  }
}

// ---------------------------------------------------------------------------
// Kernel 4: attention + output head. One block per batch row.
// ---------------------------------------------------------------------------
__global__ void __launch_bounds__(256)
k_attn(const __bf16* __restrict__ outs, const float* __restrict__ wv_W,
       const float* __restrict__ wv_b, const float* __restrict__ wu,
       const float* __restrict__ h2o_W, const float* __restrict__ h2o_b,
       float* __restrict__ out) {
  __shared__ float sWv[CA * CH];   // 40KB
  __shared__ float sS[CT];         // 4KB
  __shared__ float sRed[256];      // 1KB

  const int tid = threadIdx.x;
  const int b = blockIdx.x;
  const int lane = tid & 31;
  const int wave = tid >> 5;

  for (int i = tid; i < CA * CH; i += 256) sWv[i] = wv_W[i];
  __syncthreads();

  // Pass 1: scores s[t], wave-parallel over t, lane-parallel over a.
  for (int t = wave; t < CT; t += 8) {
    const __bf16* op = outs + ((size_t)b * CT + t) * CH;
    float part = 0.0f;
    for (int a = lane; a < CA; a += 32) {
      const float* wp = sWv + a * CH;
      float dot = wv_b[a];
#pragma unroll 8
      for (int k = 0; k < CH; ++k) dot += (float)op[k] * wp[k];
      part += tanhf(dot) * wu[a];
    }
#pragma unroll
    for (int o = 16; o > 0; o >>= 1) part += __shfl_xor(part, o, 32);
    if (lane == 0) sS[t] = part;
  }
  __syncthreads();

  // Softmax over T.
  float m = -1e30f;
  for (int t = tid; t < CT; t += 256) m = fmaxf(m, sS[t]);
  sRed[tid] = m;
  __syncthreads();
  for (int s = 128; s > 0; s >>= 1) {
    if (tid < s) sRed[tid] = fmaxf(sRed[tid], sRed[tid + s]);
    __syncthreads();
  }
  const float mx = sRed[0];
  __syncthreads();

  float sum = 0.0f;
  for (int t = tid; t < CT; t += 256) {
    float e = __expf(sS[t] - mx);
    sS[t] = e;
    sum += e;
  }
  sRed[tid] = sum;
  __syncthreads();
  for (int s = 128; s > 0; s >>= 1) {
    if (tid < s) sRed[tid] += sRed[tid + s];
    __syncthreads();
  }
  const float inv = 1.0f / sRed[0];
  __syncthreads();

  // Pass 2: context column per thread (tid == hidden index k), then head dot.
  float acc = 0.0f;
  for (int t = 0; t < CT; ++t)
    acc += sS[t] * (float)outs[((size_t)b * CT + t) * CH + tid];
  float val = acc * inv * h2o_W[tid];
  sRed[tid] = val;
  __syncthreads();
  for (int s = 128; s > 0; s >>= 1) {
    if (tid < s) sRed[tid] += sRed[tid + s];
    __syncthreads();
  }
  if (tid == 0) out[b] = sRed[0] + h2o_b[0];
}

// ---------------------------------------------------------------------------
extern "C" void kernel_launch(void* const* d_in, const int* in_sizes, int n_in,
                              void* d_out, int out_size, void* d_ws, size_t ws_size,
                              hipStream_t stream) {
  const float* x     = (const float*)d_in[0];
  const float* W_ih  = (const float*)d_in[1];
  const float* W_hh  = (const float*)d_in[2];
  const float* b_ih  = (const float*)d_in[3];
  const float* b_hh  = (const float*)d_in[4];
  const float* wv_W  = (const float*)d_in[5];
  const float* wv_b  = (const float*)d_in[6];
  const float* wu    = (const float*)d_in[7];
  const float* h2o_W = (const float*)d_in[8];
  const float* h2o_b = (const float*)d_in[9];
  float* out = (float*)d_out;

  // Workspace layout (bf16 scratch): gx [B*T*3H] | outs [B*T*H] | h double buf | barrier
  const size_t NGX = (size_t)CB * CT * CG;   // 100,663,296
  const size_t NOU = (size_t)CB * CT * CH;   //  33,554,432
  __bf16* gx   = (__bf16*)d_ws;
  __bf16* outs = gx + NGX;
  __bf16* hbuf = outs + NOU;                 // 2 * B * H
  unsigned* bar = (unsigned*)(hbuf + 2 * (size_t)CB * CH);

  // Kernel 1: input-side projections (131072x768x128 bf16 WMMA GEMM)
  const int tiles = (CB * CT / 16) * (CG / 16);   // 393216
  k_gemm_gx<<<tiles / 8, 256, 0, stream>>>(x, W_ih, b_ih, gx);

  // Kernel 2: zero h0 and the barrier counter (every call; ws not re-poisoned)
  k_scan_init<<<64, 256, 0, stream>>>(hbuf, bar);

  // Kernel 3: persistent GRU scan, TDM-staged gx, grid barrier per timestep
  k_gru_scan<<<NWG_SCAN, 256, 0, stream>>>(gx, outs, hbuf, bar, W_hh, b_hh);

  // Kernel 4: attention + output head
  k_attn<<<CB, 256, 0, stream>>>(outs, wv_W, wv_b, wu, h2o_W, h2o_b, out);
}
